// CrossAttention_83141976916044
// MI455X (gfx1250) — matrix operand order repaired and checked by reading, not compiled, not run on hardware
//
#include <hip/hip_runtime.h>
#include <hip/hip_bf16.h>

#define HIDDEN 512
#define NPOS   1024   // 32*32 spatial positions
#define BATCH  64
#define KC     32     // K-chunk per WMMA (bf16 K=32)
#define BN     64     // spatial tile width (scores kernel)
#define BND    128    // spatial tile width (context kernel)
#define CB     64     // channel block (context kernel)

typedef __attribute__((ext_vector_type(16))) __bf16 v16bf;
typedef __attribute__((ext_vector_type(8)))  __bf16 v8bf;
typedef __attribute__((ext_vector_type(2)))  __bf16 v2bf;
typedef __attribute__((ext_vector_type(8)))  float  v8f;

// pointer types matching the async-load builtin's signature
typedef int v4i_ __attribute__((vector_size(16)));
typedef __attribute__((address_space(1))) v4i_ gv4i;   // global int4
typedef __attribute__((address_space(3))) v4i_ lv4i;   // LDS int4

// ---- hardware tanh (V_TANH_F32 on gfx1250) with fallback --------------------
#if __has_builtin(__builtin_amdgcn_tanhf)
#define TANH(x) __builtin_amdgcn_tanhf(x)
#elif __has_builtin(__builtin_amdgcn_tanh_f32)
#define TANH(x) __builtin_amdgcn_tanh_f32(x)
#else
#define TANH(x) tanhf(x)
#endif

// float -> bf16, round-to-nearest-even, bit manipulation fallback
__device__ __forceinline__ __bf16 f2bf(float f) {
    unsigned u = __builtin_bit_cast(unsigned, f);
    u += 0x7FFFu + ((u >> 16) & 1u);
    unsigned short h = (unsigned short)(u >> 16);
    return __builtin_bit_cast(__bf16, h);
}

// packed 2x f32 -> 2x bf16 (V_CVT_PK_BF16_F32 on gfx1250) with fallback
__device__ __forceinline__ v2bf pk_bf(float a, float b) {
#if __has_builtin(__builtin_amdgcn_cvt_pk_bf16_f32)
    auto t = __builtin_amdgcn_cvt_pk_bf16_f32(a, b);
    return __builtin_bit_cast(v2bf, t);
#else
    v2bf r; r[0] = f2bf(a); r[1] = f2bf(b); return r;
#endif
}

// ---- async global->LDS copy (GLOBAL_LOAD_ASYNC_TO_LDS_B128, ASYNCcnt) ------
#if __has_builtin(__builtin_amdgcn_global_load_async_to_lds_b128)
#define HAVE_ASYNC_LDS 1
#endif

__device__ __forceinline__ void copy16_to_lds(const __bf16* g, __bf16* l) {
#ifdef HAVE_ASYNC_LDS
    __builtin_amdgcn_global_load_async_to_lds_b128(
        (gv4i*)(uintptr_t)g, (lv4i*)(unsigned)(uintptr_t)l, 0, 0);
#else
    *(v8bf*)l = *(const v8bf*)g;
#endif
}

__device__ __forceinline__ void async_wait_all() {
#ifdef HAVE_ASYNC_LDS
#if __has_builtin(__builtin_amdgcn_s_wait_asynccnt)
    __builtin_amdgcn_s_wait_asynccnt(0);
#else
    asm volatile("s_wait_asynccnt 0" ::: "memory");
#endif
#endif
}

// A-matrix 16x32 bf16 fragment (ISA 7.12.2):
// lane half h: elems 0..7 -> K = 8h + e ; elems 8..15 -> K = 16 + 8h + (e-8)
__device__ __forceinline__ v16bf load_fragA(const __bf16* row, int half) {
    v8bf lo = *(const v8bf*)(row + 8 * half);
    v8bf hi = *(const v8bf*)(row + 16 + 8 * half);
    v16bf r;
#pragma unroll
    for (int i = 0; i < 8; ++i) { r[i] = lo[i]; r[i + 8] = hi[i]; }
    return r;
}

// B-matrix 32x16 bf16 fragment: lane = column, same K order as A:
// lane half h: elems 0..7 -> K = 8h + e ; elems 8..15 -> K = 16 + 8h + (e-8)
__device__ __forceinline__ v16bf load_fragB(const __bf16* row, int half) {
    v8bf lo = *(const v8bf*)(row + 8 * half);
    v8bf hi = *(const v8bf*)(row + 16 + 8 * half);
    v16bf r;
#pragma unroll
    for (int i = 0; i < 8; ++i) { r[i] = lo[i]; r[i + 8] = hi[i]; }
    return r;
}

// ---------------------------------------------------------------------------
// Kernel 0: one-time fp32 -> bf16 conversion of Wk and Wv (each 512x512).
// idx 0..32767 -> Wk, 32768..65535 -> Wv ; 8 elements per thread.
// ---------------------------------------------------------------------------
__global__ void __launch_bounds__(256)
cvt_w_kernel(const float* __restrict__ Wk, const float* __restrict__ Wv,
             __bf16* __restrict__ Wkb, __bf16* __restrict__ Wvb) {
    int idx = blockIdx.x * 256 + threadIdx.x;       // 0..65535
    const float* src = (idx < 32768) ? Wk : Wv;
    __bf16*      dst = (idx < 32768) ? Wkb : Wvb;
    int e = (idx & 32767) * 8;
    float4 a = *(const float4*)(src + e);
    float4 b = *(const float4*)(src + e + 4);
    v2bf* o = (v2bf*)(dst + e);
    o[0] = pk_bf(a.x, a.y); o[1] = pk_bf(a.z, a.w);
    o[2] = pk_bf(b.x, b.y); o[3] = pk_bf(b.z, b.w);
}

// ---------------------------------------------------------------------------
// Kernel A: q[b,d] = tanh(h[b,:] . Wq[d,:] + bq[d])     (64x512x512, trivial)
// ---------------------------------------------------------------------------
__global__ void __launch_bounds__(256)
query_kernel(const float* __restrict__ h, const float* __restrict__ Wq,
             const float* __restrict__ bq, float* __restrict__ qout) {
    int idx = blockIdx.x * 256 + threadIdx.x;      // 0..32767
    int b = idx >> 9;
    int d = idx & (HIDDEN - 1);
    const float4* hv = (const float4*)(h + (size_t)b * HIDDEN);
    const float4* wv = (const float4*)(Wq + (size_t)d * HIDDEN);
    float s = 0.f;
#pragma unroll 4
    for (int i = 0; i < HIDDEN / 4; ++i) {
        float4 a = hv[i], w = wv[i];
        s += a.x * w.x + a.y * w.y + a.z * w.z + a.w * w.w;
    }
    qout[idx] = TANH(s + bq[d]);
}

// ---------------------------------------------------------------------------
// Kernel B: scores[b,n] = sum_d q[b,d] * tanh( (Wk . img_b)[d,n] + bk[d] )
// Keys never leave registers. Block = 512 thr (16 waves), grid (N/BN, B).
// Wave w owns d-rows [32w, 32w+32) x the full 64-wide n tile.
// ---------------------------------------------------------------------------
__global__ void __launch_bounds__(512)
scores_kernel(const float* __restrict__ img,   // [B,512,1024] fp32
              const __bf16* __restrict__ Wkb,  // [512,512] bf16
              const float* __restrict__ bk,    // [512]
              const float* __restrict__ qv,    // [B,512]
              float* __restrict__ scores) {    // [B,1024]
    __shared__ __bf16 lA[HIDDEN * KC];   // Wk[d][k-chunk]   32 KB
    __shared__ __bf16 lB[BN * KC];       // img^T [n][k]      4 KB
    __shared__ float  sred[BN];

    const int b    = blockIdx.y;
    const int n0   = blockIdx.x * BN;
    const int tid  = threadIdx.x;
    const int lane = tid & 31;
    const int wave = tid >> 5;           // 0..15
    const int half = lane >> 4;
    const int l16  = lane & 15;

    if (tid < BN) sred[tid] = 0.f;

    const v8f vz = {0.f, 0.f, 0.f, 0.f, 0.f, 0.f, 0.f, 0.f};
    v8f acc[2][4];
#pragma unroll
    for (int mt = 0; mt < 2; ++mt)
#pragma unroll
        for (int nt = 0; nt < 4; ++nt) acc[mt][nt] = vz;

    const float* imgb = img + (size_t)b * HIDDEN * NPOS;

    for (int k0 = 0; k0 < HIDDEN; k0 += KC) {
        __syncthreads();
        // stage A (raw bf16 async copy): one Wk row per thread, 32 elems = 64 B
        {
            const __bf16* g = Wkb + (size_t)tid * HIDDEN + k0;
            __bf16* l = lA + tid * KC;
#pragma unroll
            for (int i = 0; i < 4; ++i)
                copy16_to_lds(g + 8 * i, l + 8 * i);
        }
        // stage B (transpose [k][n] -> [n][k], fp32 -> bf16): 4 elems/thread
        {
            int k = tid >> 4;
            int n = (tid & 15) * 4;
            float4 v = *(const float4*)(imgb + (size_t)(k0 + k) * NPOS + n0 + n);
            lB[(n+0)*KC + k] = f2bf(v.x);
            lB[(n+1)*KC + k] = f2bf(v.y);
            lB[(n+2)*KC + k] = f2bf(v.z);
            lB[(n+3)*KC + k] = f2bf(v.w);
            if (k0 + KC < HIDDEN)
                __builtin_prefetch(imgb + (size_t)(k0 + KC + k) * NPOS + n0 + n, 0, 1);
        }
        async_wait_all();
        __syncthreads();

        v16bf fb[4];
#pragma unroll
        for (int nt = 0; nt < 4; ++nt)
            fb[nt] = load_fragB(lB + (nt * 16 + l16) * KC, half);
#pragma unroll
        for (int mt = 0; mt < 2; ++mt) {
            v16bf fa = load_fragA(lA + (wave * 32 + mt * 16 + l16) * KC, half);
#pragma unroll
            for (int nt = 0; nt < 4; ++nt)
                acc[mt][nt] = __builtin_amdgcn_wmma_f32_16x16x32_bf16(
                    false, fa, false, fb[nt], (short)0, acc[mt][nt], false, false);
        }
    }
    __syncthreads();

    // epilogue: tanh(+bk) * q[d], reduce over d
    float part[4] = {0.f, 0.f, 0.f, 0.f};
#pragma unroll
    for (int mt = 0; mt < 2; ++mt) {
#pragma unroll
        for (int r = 0; r < 8; ++r) {
            int d = wave * 32 + mt * 16 + r + 8 * half;
            float bkd = bk[d];
            float qd  = qv[(size_t)b * HIDDEN + d];
#pragma unroll
            for (int nt = 0; nt < 4; ++nt)
                part[nt] += TANH(acc[mt][nt][r] + bkd) * qd;
        }
    }
#pragma unroll
    for (int nt = 0; nt < 4; ++nt) {
        float o = __shfl_down(part[nt], 16, 32);   // fold lanes 16-31 onto 0-15
        if (half == 0) atomicAdd(&sred[nt * 16 + l16], part[nt] + o);
    }
    __syncthreads();
    if (tid < BN) scores[(size_t)b * NPOS + n0 + tid] = sred[tid];
}

// ---------------------------------------------------------------------------
// Kernel C: softmax over N per batch. One 256-thread block per batch.
// ---------------------------------------------------------------------------
__global__ void __launch_bounds__(256)
softmax_kernel(const float* __restrict__ scores, float* __restrict__ wts) {
    __shared__ float red[256];
    const int b = blockIdx.x, tid = threadIdx.x;
    const float* s = scores + (size_t)b * NPOS;
    float* w = wts + (size_t)b * NPOS;

    float mx = -3.4e38f;
    for (int i = tid; i < NPOS; i += 256) mx = fmaxf(mx, s[i]);
    red[tid] = mx; __syncthreads();
    for (int off = 128; off > 0; off >>= 1) {
        if (tid < off) red[tid] = fmaxf(red[tid], red[tid + off]);
        __syncthreads();
    }
    float m = red[0]; __syncthreads();

    float sum = 0.f;
    for (int i = tid; i < NPOS; i += 256) {
        float e = expf(s[i] - m);
        w[i] = e; sum += e;
    }
    red[tid] = sum; __syncthreads();
    for (int off = 128; off > 0; off >>= 1) {
        if (tid < off) red[tid] += red[tid + off];
        __syncthreads();
    }
    float inv = 1.f / red[0];
    for (int i = tid; i < NPOS; i += 256) w[i] *= inv;
}

// ---------------------------------------------------------------------------
// Kernel D: context[b,c] = sum_n w[b,n] * tanh( (Wv . img_b)[c,n] + bv[c] )
// Values never leave registers. Block = 256 thr (8 waves), grid (8 cblk, B).
// Wave w: mi = w&3 (one 16-row M tile), ni = w>>2 (64 n cols -> 4 accum tiles,
// A fragment reused across 4 WMMAs per K-step).
// ---------------------------------------------------------------------------
__global__ void __launch_bounds__(256)
context_kernel(const float* __restrict__ img, const __bf16* __restrict__ Wvb,
               const float* __restrict__ bv, const float* __restrict__ wts,
               float* __restrict__ out) {      // [B,512]
    __shared__ __bf16 lA[CB * KC];    // Wv[c-block][k-chunk]  4 KB
    __shared__ __bf16 lB[BND * KC];   // img^T [n][k]          8 KB
    __shared__ float  ctx[CB];

    const int b    = blockIdx.y;
    const int c0   = blockIdx.x * CB;
    const int tid  = threadIdx.x;
    const int lane = tid & 31;
    const int wave = tid >> 5;   // 0..7
    const int mi   = wave & 3;   // M tile (16 rows)
    const int ni   = wave >> 2;  // 0..1 -> 64-wide n group
    const int half = lane >> 4;
    const int l16  = lane & 15;

    if (tid < CB) ctx[tid] = 0.f;

    const float* imgb = img + (size_t)b * HIDDEN * NPOS;
    const float* wb   = wts + (size_t)b * NPOS;
    const v8f vz = {0.f, 0.f, 0.f, 0.f, 0.f, 0.f, 0.f, 0.f};

    float ctxacc[4][8];
#pragma unroll
    for (int nt = 0; nt < 4; ++nt)
#pragma unroll
        for (int r = 0; r < 8; ++r) ctxacc[nt][r] = 0.f;

    for (int n0 = 0; n0 < NPOS; n0 += BND) {
        v8f acc[4] = {vz, vz, vz, vz};
        for (int k0 = 0; k0 < HIDDEN; k0 += KC) {
            __syncthreads();
            // stage A (raw bf16 async copy): 8 elems/thread = one B128
            {
                int m  = tid >> 2;
                int kq = (tid & 3) * 8;
                copy16_to_lds(Wvb + (size_t)(c0 + m) * HIDDEN + k0 + kq,
                              lA + m * KC + kq);
            }
            // stage B (transpose): k = tid>>3, n16 = (tid&7)*16; 16 elems/thread
            {
                int k   = tid >> 3;
                int n16 = (tid & 7) * 16;
                const float4* src = (const float4*)(imgb + (size_t)(k0 + k) * NPOS + n0 + n16);
#pragma unroll
                for (int j = 0; j < 4; ++j) {
                    float4 v = src[j];
                    lB[(n16 + 4*j + 0)*KC + k] = f2bf(v.x);
                    lB[(n16 + 4*j + 1)*KC + k] = f2bf(v.y);
                    lB[(n16 + 4*j + 2)*KC + k] = f2bf(v.z);
                    lB[(n16 + 4*j + 3)*KC + k] = f2bf(v.w);
                }
            }
            async_wait_all();
            __syncthreads();

            v16bf fa = load_fragA(lA + (mi * 16 + l16) * KC, half);
#pragma unroll
            for (int nt = 0; nt < 4; ++nt) {
                v16bf fb = load_fragB(lB + (ni * 64 + nt * 16 + l16) * KC, half);
                acc[nt] = __builtin_amdgcn_wmma_f32_16x16x32_bf16(
                    false, fa, false, fb, (short)0, acc[nt], false, false);
            }
        }
        // per-n-tile epilogue: tanh(+bv) * weight, fold into running context
#pragma unroll
        for (int nt = 0; nt < 4; ++nt) {
            float w = wb[n0 + ni * 64 + nt * 16 + l16];
#pragma unroll
            for (int r = 0; r < 8; ++r) {
                float bvd = bv[c0 + mi * 16 + r + 8 * half];
                ctxacc[nt][r] += w * TANH(acc[nt][r] + bvd);
            }
        }
    }
    __syncthreads();   // ctx[] zero-init visible

    // reduce across the 16 lanes of each half, then across waves via LDS atomics
#pragma unroll
    for (int nt = 0; nt < 4; ++nt) {
#pragma unroll
        for (int r = 0; r < 8; ++r) {
            float s = ctxacc[nt][r];
            s += __shfl_xor(s, 1, 16);
            s += __shfl_xor(s, 2, 16);
            s += __shfl_xor(s, 4, 16);
            s += __shfl_xor(s, 8, 16);
            if (l16 == 0) atomicAdd(&ctx[mi * 16 + r + 8 * half], s);
        }
    }
    __syncthreads();
    if (tid < CB) out[(size_t)b * HIDDEN + c0 + tid] = ctx[tid];
}

// ---------------------------------------------------------------------------
extern "C" void kernel_launch(void* const* d_in, const int* in_sizes, int n_in,
                              void* d_out, int out_size, void* d_ws, size_t ws_size,
                              hipStream_t stream) {
    (void)in_sizes; (void)n_in; (void)out_size; (void)ws_size;
    const float* img = (const float*)d_in[0];   // [64,512,32,32]
    const float* h   = (const float*)d_in[1];   // [64,512]
    const float* Wq  = (const float*)d_in[2];
    const float* bq  = (const float*)d_in[3];
    const float* Wk  = (const float*)d_in[4];
    const float* bk  = (const float*)d_in[5];
    const float* Wv  = (const float*)d_in[6];
    const float* bv  = (const float*)d_in[7];
    float* out = (float*)d_out;                 // [64,512]

    float*  qbuf   = (float*)d_ws;                 // 64*512 f32
    float*  scores = qbuf + BATCH * HIDDEN;        // 64*1024 f32
    float*  wts    = scores + BATCH * NPOS;        // 64*1024 f32
    __bf16* Wkb    = (__bf16*)(wts + BATCH * NPOS);        // 512*512 bf16
    __bf16* Wvb    = Wkb + HIDDEN * HIDDEN;                // 512*512 bf16

    cvt_w_kernel<<<256, 256, 0, stream>>>(Wk, Wv, Wkb, Wvb);
    query_kernel<<<(BATCH * HIDDEN) / 256, 256, 0, stream>>>(h, Wq, bq, qbuf);
    scores_kernel<<<dim3(NPOS / BN, BATCH), 512, 0, stream>>>(img, Wkb, bk, qbuf, scores);
    softmax_kernel<<<BATCH, 256, 0, stream>>>(scores, wts);
    context_kernel<<<dim3(HIDDEN / CB, BATCH), 256, 0, stream>>>(img, Wvb, bv, wts, out);
}
